// Model_28475633172566
// MI455X (gfx1250) — matrix operand order repaired and checked by reading, not compiled, not run on hardware
//
#include <hip/hip_runtime.h>
#include <hip/hip_fp16.h>
#include <math.h>

typedef __attribute__((ext_vector_type(16))) _Float16 v16h;
typedef __attribute__((ext_vector_type(8)))  float    v8f;

#define B_  16
#define R_  5
#define C_  200
#define N_  201
#define H_  128
#define BN  (B_*N_)       // 3216 rows
#define MT  (BN/16)       // 201 row tiles (exact)

__device__ __forceinline__ float fast_tanh(float x){
#if __has_builtin(__builtin_amdgcn_tanhf)
  return __builtin_amdgcn_tanhf(x);
#else
  return tanhf(x);
#endif
}

// ---------------------------------------------------------------------------
// Pack the three HxH weight matrices into the wave32 WMMA B-operand layout:
// B[k][n] = W[n][k];  per (ntile t, kstep kk): lane holds 16 halves,
// col n = t*16 + (lane&15), k = kk*32 + (lane>=16 ? 16 : 0) + j.
// Stored as [w][t][kk][lane][j] -> each lane later does one contiguous v16h load.
// ---------------------------------------------------------------------------
__global__ void k_pack(const float* __restrict__ Wl1, const float* __restrict__ Wx2,
                       const float* __restrict__ Wl2, _Float16* __restrict__ P){
  const float* Ws[3] = {Wl1, Wx2, Wl2};
  int total = 3*16384;
  for (int idx = blockIdx.x*blockDim.x + threadIdx.x; idx < total;
       idx += gridDim.x*blockDim.x){
    int w    = idx >> 14;
    int r    = idx & 16383;
    int j    = r & 15;
    int lane = (r >> 4) & 31;
    int kk   = (r >> 9) & 3;
    int t    = (r >> 11) & 7;
    int n = t*16 + (lane & 15);
    int k = kk*32 + ((lane >= 16) ? 16 : 0) + j;
    P[idx] = (_Float16)Ws[w][n*H_ + k];
  }
}

// ---------------------------------------------------------------------------
// x = [xa(2), x_b(2), coord(2), avail(1)] ; xW[b,n,h] = sum_f x[f]*Wx1[h,f]
// ---------------------------------------------------------------------------
__global__ void k_prep(const float* __restrict__ action, const float* __restrict__ aprev,
                       const float* __restrict__ x_a, const float* __restrict__ x_b,
                       const float* __restrict__ coord, const float* __restrict__ avail,
                       const float* __restrict__ Wx1, float* __restrict__ xW){
  int bn = blockIdx.x; int b = bn / N_; int n = bn % N_;
  float xv[7];
  float xa0 = 0.f, xa1 = 0.f;
  #pragma unroll
  for (int r = 0; r < R_; r++){
    int i = (b*R_ + r)*N_ + n;
    float s = aprev[i] + action[i];
    xa0 += s * x_a[i*2+0];
    xa1 += s * x_a[i*2+1];
  }
  xv[0]=xa0; xv[1]=xa1;
  xv[2]=x_b[(b*N_+n)*2+0];  xv[3]=x_b[(b*N_+n)*2+1];
  xv[4]=coord[(b*N_+n)*2+0];xv[5]=coord[(b*N_+n)*2+1];
  xv[6]=avail[b*N_+n];
  int h = threadIdx.x;
  float acc = 0.f;
  #pragma unroll
  for (int f = 0; f < 7; f++) acc += xv[f]*Wx1[h*7+f];
  xW[(size_t)bn*H_ + h] = acc;
}

// ---------------------------------------------------------------------------
// Edge MLP + masked softmax over n (one block per (b,c)); writes transposed
// pT[b][n][c] and eT[b][n][c] so the message kernel reads contiguous in c.
// ---------------------------------------------------------------------------
__global__ void k_edge(const float* __restrict__ edge, const float* __restrict__ avail,
                       const float* __restrict__ W1p, const float* __restrict__ W2p,
                       float* __restrict__ pT, float* __restrict__ eT){
  __shared__ float sW1[3*H_];
  __shared__ float sW2[H_];
  __shared__ float red[256];
  int bc = blockIdx.x; int b = bc / C_; int c = bc % C_;
  int tid = threadIdx.x;
  for (int i = tid; i < 3*H_; i += 256) sW1[i] = W1p[i];
  if (tid < H_) sW2[tid] = W2p[tid];
  __syncthreads();

  int n = tid;
  float logit = -3.0e38f;
  float e0 = 0.f;
  if (n < N_){
    const float* ep = edge + ((size_t)(b*C_ + c)*N_ + n)*3;
    e0 = ep[0]; float e1 = ep[1], e2 = ep[2];
    float acc = 0.f;
    #pragma unroll 4
    for (int h = 0; h < H_; h++){
      float h1 = e0*sW1[3*h] + e1*sW1[3*h+1] + e2*sW1[3*h+2];
      h1 = h1 > 0.f ? h1 : 0.f;
      acc += sW2[h]*h1;
    }
    float h2v = (acc > 0.f ? acc : 0.f) * 0.5f;   // /TAU
    float av = avail[b*N_ + n];
    float mask = (n == c ? 0.f : 1.f)*av;
    logit = h2v*mask - (1.f - mask)*1.0e10f;
  }
  red[tid] = logit; __syncthreads();
  for (int s = 128; s > 0; s >>= 1){
    if (tid < s) red[tid] = fmaxf(red[tid], red[tid+s]);
    __syncthreads();
  }
  float mx = red[0]; __syncthreads();
  float ex = (n < N_) ? __expf(logit - mx) : 0.f;
  red[tid] = ex; __syncthreads();
  for (int s = 128; s > 0; s >>= 1){
    if (tid < s) red[tid] += red[tid+s];
    __syncthreads();
  }
  float inv = 1.f / red[0];
  if (n < N_){
    float scale = avail[b*N_ + c];   // avail[:,0,:-1] row scale
    pT[(size_t)(b*N_ + n)*C_ + c] = ex*inv*scale;
    eT[(size_t)(b*N_ + n)*C_ + c] = e0;
  }
}

// state init: u = SIGMA * noise
__global__ void k_scale(const float* __restrict__ src, float* __restrict__ dst, int count){
  for (int i = blockIdx.x*blockDim.x + threadIdx.x; i < count; i += gridDim.x*blockDim.x)
    dst[i] = 0.001f*src[i];
}

// ---------------------------------------------------------------------------
// l[b,n,h] = sum_c p[b,c,n] * tanh(e[b,c,n]*We[h]) * state[b,c,h]
// One block per (b,n), thread = h. p/e rows cached in LDS; state rows L2-hot.
// ---------------------------------------------------------------------------
__global__ void k_msg(const float* __restrict__ pT, const float* __restrict__ eT,
                      const float* __restrict__ We, const float* __restrict__ U,
                      float* __restrict__ L){
  __shared__ float sa[C_];
  __shared__ float se[C_];
  int bn = blockIdx.x; int b = bn / N_;
  int h = threadIdx.x;
  for (int i = h; i < C_; i += H_){
    sa[i] = pT[(size_t)bn*C_ + i];
    se[i] = eT[(size_t)bn*C_ + i];
  }
  __syncthreads();
  float w = We[h];
  const float* ub = U + (size_t)b*N_*H_ + h;
  float acc = 0.f;
  #pragma unroll 4
  for (int c = 0; c < C_; c++){
    acc += sa[c]*fast_tanh(se[c]*w)*ub[(size_t)c*H_];
  }
  L[(size_t)bn*H_ + h] = acc;
}

// ---------------------------------------------------------------------------
// Out(3216x128) = [relu]( A(3216x128) * W^T + bias ) via v_wmma_f32_16x16x32_f16
// One wave per 16-row stripe; 8 N-tiles x 4 K-steps = 32 WMMAs. EXEC all ones.
// A gathered in contiguous k order, matching the B pack:
//   k(j,lane) = (lane>=16?16:0) + j
// B loaded as one contiguous v16h from the pre-packed buffer.
// ---------------------------------------------------------------------------
__global__ void k_gemm(const float* __restrict__ A, const _Float16* __restrict__ Bp,
                       const float* __restrict__ bias, float* __restrict__ Out,
                       int apply_relu){
  int lane = threadIdx.x;
  int row0 = blockIdx.x*16;
  v8f c[8];
  #pragma unroll
  for (int t = 0; t < 8; t++) c[t] = (v8f){0,0,0,0,0,0,0,0};

  int m    = lane & 15;
  int khiA = (lane >= 16) ? 16 : 0;
  const v16h* bptr = (const v16h*)Bp;

  #pragma unroll
  for (int kk = 0; kk < 4; kk++){
    const float* arow = A + (size_t)(row0 + m)*H_ + kk*32;
    v16h a;
    #pragma unroll
    for (int j = 0; j < 16; j++){
      int k = khiA + j;
      a[j] = (_Float16)arow[k];
    }
    #pragma unroll
    for (int t = 0; t < 8; t++){
      v16h bm = bptr[(t*4 + kk)*32 + lane];
      c[t] = __builtin_amdgcn_wmma_f32_16x16x32_f16(
                 false, a, false, bm, (short)0, c[t], false, false);
    }
  }

  int mrow = 8*(lane >> 4);   // D layout: lanes 0-15 -> M=r, lanes 16-31 -> M=r+8
  #pragma unroll
  for (int t = 0; t < 8; t++){
    int col = t*16 + (lane & 15);
    #pragma unroll
    for (int r = 0; r < 8; r++){
      int row = row0 + mrow + r;
      float v = c[t][r];
      if (bias) v += bias[(size_t)row*H_ + col];
      if (apply_relu) v = v > 0.f ? v : 0.f;
      Out[(size_t)row*H_ + col] = v;
    }
  }
}

// ---------------------------------------------------------------------------
// Q[b] = sum_h ( (sum_n g*av)*WQ[h] + (sum_n g*(1-av))*WQ[H+h] ) + bQ
// ---------------------------------------------------------------------------
__global__ void k_q(const float* __restrict__ gamma, const float* __restrict__ avail,
                    const float* __restrict__ WQ, const float* __restrict__ bQ,
                    float* __restrict__ out){
  __shared__ float red[H_];
  int b = blockIdx.x; int h = threadIdx.x;
  float rem = 0.f, don = 0.f;
  for (int n = 0; n < N_; n++){
    float g  = gamma[(size_t)(b*N_ + n)*H_ + h];
    float av = avail[b*N_ + n];
    rem += g*av;
    don += g*(1.f - av);
  }
  red[h] = rem*WQ[h] + don*WQ[H_ + h];
  __syncthreads();
  for (int s = 64; s > 0; s >>= 1){
    if (h < s) red[h] += red[h+s];
    __syncthreads();
  }
  if (h == 0) out[b] = red[0] + bQ[0];
}

extern "C" void kernel_launch(void* const* d_in, const int* in_sizes, int n_in,
                              void* d_out, int out_size, void* d_ws, size_t ws_size,
                              hipStream_t stream){
  const float* action = (const float*)d_in[0];
  const float* aprev  = (const float*)d_in[1];
  const float* x_a    = (const float*)d_in[2];
  const float* x_b    = (const float*)d_in[3];
  const float* coord  = (const float*)d_in[4];
  const float* edge   = (const float*)d_in[5];
  /* d_in[6] presence_prev: unused by reference */
  const float* avail  = (const float*)d_in[7];
  const float* noise_u= (const float*)d_in[8];
  const float* noise_g= (const float*)d_in[9];
  const float* W1p = (const float*)d_in[10];
  const float* W2p = (const float*)d_in[11];
  const float* Wx1 = (const float*)d_in[12];
  const float* We1 = (const float*)d_in[13];
  const float* Wl1 = (const float*)d_in[14];
  const float* Wx2 = (const float*)d_in[15];
  const float* We2 = (const float*)d_in[16];
  const float* Wl2 = (const float*)d_in[17];
  const float* WQ  = (const float*)d_in[18];
  const float* bQ  = (const float*)d_in[19];
  float* out = (float*)d_out;

  char* ws = (char*)d_ws;
  size_t off = 0;
  auto alloc = [&](size_t bytes)->void*{
    void* p = ws + off;
    off += (bytes + 255) & ~(size_t)255;
    return p;
  };
  float* xW   = (float*)alloc((size_t)BN*H_*4);
  float* pT   = (float*)alloc((size_t)B_*N_*C_*4);
  float* eT   = (float*)alloc((size_t)B_*N_*C_*4);
  float* u    = (float*)alloc((size_t)BN*H_*4);
  float* gam  = (float*)alloc((size_t)BN*H_*4);
  float* lbuf = (float*)alloc((size_t)BN*H_*4);
  float* uW   = (float*)alloc((size_t)BN*H_*4);
  _Float16* Bp    = (_Float16*)alloc((size_t)3*16384*2);
  _Float16* Bp_l1 = Bp;
  _Float16* Bp_x2 = Bp + 16384;
  _Float16* Bp_l2 = Bp + 32768;

  k_pack<<<48, 256, 0, stream>>>(Wl1, Wx2, Wl2, Bp);
  k_prep<<<BN, H_, 0, stream>>>(action, aprev, x_a, x_b, coord, avail, Wx1, xW);
  k_edge<<<B_*C_, 256, 0, stream>>>(edge, avail, W1p, W2p, pT, eT);
  k_scale<<<512, 256, 0, stream>>>(noise_u, u, BN*H_);
  k_scale<<<512, 256, 0, stream>>>(noise_g, gam, BN*H_);

  for (int i = 0; i < 4; i++){                   // u recurrence (T1)
    k_msg<<<BN, H_, 0, stream>>>(pT, eT, We1, u, lbuf);
    k_gemm<<<MT, 32, 0, stream>>>(lbuf, Bp_l1, xW, u, 1);
  }
  k_gemm<<<MT, 32, 0, stream>>>(u, Bp_x2, nullptr, uW, 0);   // uW = u * Wx2^T
  for (int i = 0; i < 4; i++){                   // gamma recurrence (T2)
    k_msg<<<BN, H_, 0, stream>>>(pT, eT, We2, gam, lbuf);
    k_gemm<<<MT, 32, 0, stream>>>(lbuf, Bp_l2, uW, gam, 1);
  }
  k_q<<<B_, H_, 0, stream>>>(gam, avail, WQ, bQ, out);
}
